// MultibandFrameAttention_30099130811183
// MI455X (gfx1250) — compile-verified, matrix operand order checked by reading, not run on hardware
//
#include <hip/hip_runtime.h>

// ---------------------------------------------------------------------------
// MultibandFrameAttention for MI455X (gfx1250, wave32, WMMA bf16 + TDM)
// ---------------------------------------------------------------------------
// B=4, W=1024, C=1024, H=16, D=64, K(conv)=3
//
// Pipeline:
//   cast x / qW / kW / vW / oW -> bf16 ; er -> erT bf16
//   lin = x @ W.T + b            (WMMA bf16 GEMM, 64x64 tile/wave, f32 acc)
//   q/k = causal dwconv(lin)     -> bf16 (B,H,W,D) ; v -> bf16 (B,H,D,W)
//   flash attention w/ rel-pos bias p = (q @ er)^T  (WMMA; K/V tiles staged
//   into LDS by the Tensor Data Mover, double-buffered on TENSORcnt)
//   out = o @ oW.T + ob          (WMMA GEMM, fp32 out)
// ---------------------------------------------------------------------------

constexpr int BB = 4;
constexpr int WW = 1024;
constexpr int CC = 1024;
constexpr int HH = 16;
constexpr int DD = 64;          // head dim
constexpr int KK = 3;           // conv kernel
constexpr int NN = BB * WW;     // 4096 rows for the big GEMMs

typedef __bf16 bf16_t;
typedef __attribute__((ext_vector_type(16))) __bf16 v16bf;
typedef __attribute__((ext_vector_type(8)))  __bf16 v8bf;
typedef __attribute__((ext_vector_type(8)))  float  v8f;
typedef __attribute__((ext_vector_type(4)))  unsigned int u32x4;
typedef __attribute__((ext_vector_type(8)))  unsigned int u32x8;

// float -> bf16 round-to-nearest-even
__device__ __forceinline__ bf16_t f2bf(float f) {
    union { float f; unsigned u; } uf; uf.f = f;
    unsigned r = uf.u + 0x7FFFu + ((uf.u >> 16) & 1u);
    unsigned short hs = (unsigned short)(r >> 16);
    union { unsigned short s; bf16_t b; } ub; ub.s = hs;
    return ub.b;
}

// Load one 16x32 bf16 WMMA fragment from a row-major matrix whose rows are
// the M (for A) or N (for transposed-B) index and whose columns are K.
// Per ISA 16-bit layout: lanes 0-15 hold K 0..7 (v0-3) and 16..23 (v4-7),
// lanes 16-31 hold K 8..15 and 24..31 -> two contiguous b128 loads per lane.
__device__ __forceinline__ v16bf load_frag(const bf16_t* base, int stride, int lane) {
    int row = lane & 15;
    int kh  = (lane >> 4) * 8;
    const bf16_t* p = base + (size_t)row * stride + kh;
    union { v16bf v; v8bf h[2]; } u;
    u.h[0] = *(const v8bf*)(p);
    u.h[1] = *(const v8bf*)(p + 16);
    return u.v;
}

__device__ __forceinline__ v8f wmma_bf16(v16bf a, v16bf b, v8f c) {
    return __builtin_amdgcn_wmma_f32_16x16x32_bf16(false, a, false, b,
                                                   (short)0, c, false, false);
}

// ---------------------------------------------------------------------------
// Tensor Data Mover: async 2D tile load global -> LDS (ISA 08, D# §8.3/8.4).
// 2-group descriptor (2D tensor, VADDR2/3 = NULL): group0 = {flags, lds_addr,
// global_addr lo, global_addr hi | type=2}; group1 packs data_size=2B, tensor
// dims, tile dims and dim0 stride. Tracked with TENSORcnt.
// ---------------------------------------------------------------------------
__device__ __forceinline__ void tdm_load_2d(unsigned lds_off, const void* gaddr,
                                            unsigned tile_d0, unsigned tile_d1,
                                            unsigned tensor_d0, unsigned tensor_d1,
                                            unsigned stride_d0) {
    unsigned long long ga = (unsigned long long)(uintptr_t)gaddr;
    u32x4 g0;
    g0[0] = 1u;                                               // count=1 (valid D#)
    g0[1] = lds_off;                                          // lds_addr (bytes)
    g0[2] = (unsigned)(ga & 0xFFFFFFFFu);                     // global_addr[31:0]
    g0[3] = (unsigned)((ga >> 32) & 0x01FFFFFFu) | (2u << 30);// addr[56:32]|type=2
    u32x8 g1;
    g1[0] = 1u << 16;                                         // data_size=1 -> 2B
    g1[1] = (tensor_d0 & 0xFFFFu) << 16;                      // tensor_dim0 lo
    g1[2] = ((tensor_d0 >> 16) & 0xFFFFu) | ((tensor_d1 & 0xFFFFu) << 16);
    g1[3] = ((tensor_d1 >> 16) & 0xFFFFu) | ((tile_d0 & 0xFFFFu) << 16);
    g1[4] = (tile_d1 & 0xFFFFu);                              // tile_dim1, dim2=0
    g1[5] = stride_d0;                                        // dim0 stride lo
    g1[6] = 0u;
    g1[7] = 0u;
    asm volatile("tensor_load_to_lds %0, %1"
                 :: "s"(g0), "s"(g1)
                 : "memory");
}

// ---------------------------------------------------------------------------
// elementwise casts
// ---------------------------------------------------------------------------
__global__ void cast_f32_bf16_kernel(const float* __restrict__ in,
                                     bf16_t* __restrict__ out, int count) {
    int i = blockIdx.x * blockDim.x + threadIdx.x;
    int stride = gridDim.x * blockDim.x;
    for (; i < count; i += stride) out[i] = f2bf(in[i]);
}

// er (D,W) fp32 -> erT (W,D) bf16
__global__ void transpose_er_kernel(const float* __restrict__ er,
                                    bf16_t* __restrict__ erT) {
    int i = blockIdx.x * blockDim.x + threadIdx.x;
    if (i < DD * WW) {
        int d = i / WW, w = i % WW;
        erT[(size_t)w * DD + d] = f2bf(er[i]);
    }
}

// ---------------------------------------------------------------------------
// GEMM: Cout[n,o] = sum_c A[n,c] * Wt[o,c] + bias[o]   (A,Wt bf16; out f32)
// 64x64 output tile per wave: 4 A-frags x 4 B-frags -> 16 WMMA per K-step
// (2 WMMA per b128 fragment load). global_prefetch_b8 pulls the next K strip.
// ---------------------------------------------------------------------------
__global__ __launch_bounds__(256)
void gemm_bf16_kernel(const bf16_t* __restrict__ A, const bf16_t* __restrict__ Wt,
                      const float* __restrict__ bias, float* __restrict__ Cout,
                      int Nrows, int Ncols, int Kdim) {
    int lane  = threadIdx.x & 31;
    int wpb   = blockDim.x >> 5;
    int tile  = blockIdx.x * wpb + (threadIdx.x >> 5);
    int tcols = Ncols >> 6;                     // 64-wide N tiles
    int trows = Nrows >> 6;                     // 64-tall M tiles
    if (tile >= trows * tcols) return;          // wave-uniform guard
    int ti = tile / tcols, tj = tile % tcols;

    const bf16_t* Ab = A  + (size_t)ti * 64 * Kdim;
    const bf16_t* Bb = Wt + (size_t)tj * 64 * Kdim;

    v8f acc[4][4] = {};
    for (int kk = 0; kk < Kdim; kk += 32) {
        // prefetch next K strip (speculative; dropped silently if OOB)
        __builtin_prefetch(Ab + (size_t)lane * Kdim + kk + 64, 0, 1);
        __builtin_prefetch(Bb + (size_t)lane * Kdim + kk + 64, 0, 1);

        v16bf af[4], bf[4];
#pragma unroll
        for (int i = 0; i < 4; ++i)
            af[i] = load_frag(Ab + (size_t)i * 16 * Kdim + kk, Kdim, lane);
#pragma unroll
        for (int j = 0; j < 4; ++j)
            bf[j] = load_frag(Bb + (size_t)j * 16 * Kdim + kk, Kdim, lane);
#pragma unroll
        for (int i = 0; i < 4; ++i)
#pragma unroll
            for (int j = 0; j < 4; ++j)
                acc[i][j] = wmma_bf16(af[i], bf[j], acc[i][j]);
    }

    int n  = lane & 15;
    int lh = lane >> 4;
#pragma unroll
    for (int j = 0; j < 4; ++j) {
        float bv = bias[tj * 64 + j * 16 + n];
#pragma unroll
        for (int i = 0; i < 4; ++i) {
#pragma unroll
            for (int r = 0; r < 8; ++r) {
                int m = i * 16 + r + 8 * lh;
                Cout[(size_t)(ti * 64 + m) * Ncols + tj * 64 + j * 16 + n] =
                    acc[i][j][r] + bv;
            }
        }
    }
}

// ---------------------------------------------------------------------------
// Causal depthwise conv (K=3, left pad 2) + bias, with per-head re-layout.
//   transposeV == 0 : out[b,h,w,d]  (for q, k)
//   transposeV == 1 : out[b,h,d,w]  (for v -> contiguous B-operand in AV)
// ---------------------------------------------------------------------------
__global__ void conv_dw_kernel(const float* __restrict__ lin,
                               const float* __restrict__ cw,
                               const float* __restrict__ cb,
                               bf16_t* __restrict__ out, int transposeV) {
    int idx = blockIdx.x * blockDim.x + threadIdx.x;
    int stride = gridDim.x * blockDim.x;
    for (; idx < NN * CC; idx += stride) {
        int c = idx % CC;
        int w = (idx / CC) % WW;
        int b = idx / (CC * WW);
        float s = cb[c];
#pragma unroll
        for (int t = 0; t < KK; ++t) {
            int wi = w + t - (KK - 1);
            if (wi >= 0) s += lin[((size_t)b * WW + wi) * CC + c] * cw[c * KK + t];
        }
        int h = c >> 6;          // c / D
        int d = c & (DD - 1);    // c % D
        size_t o = transposeV
            ? (((size_t)b * HH + h) * DD + d) * WW + w
            : (((size_t)b * HH + h) * WW + w) * DD + d;
        out[o] = f2bf(s);
    }
}

// ---------------------------------------------------------------------------
// Flash attention, one wave per (b, h, 16-row block).
//   scores s[i,j] = ( q_i . k_j  +  q_j . er[:, i] ) / sqrt(C)
// J-tiles of 32 columns: 4 WMMA (QK^T) + 4 WMMA (rel-pos bias) + 4 WMMA (AV).
// K (32xD) and V (Dx32) tiles are staged into LDS by the Tensor Data Mover,
// double-buffered: issue next tiles, s_wait_tensorcnt 2 -> consume current.
// Online softmax; probs round-trip LDS as bf16 to become an A fragment.
// ---------------------------------------------------------------------------
__global__ __launch_bounds__(32)
void attn_kernel(const bf16_t* __restrict__ q,   // (B,H,W,D)
                 const bf16_t* __restrict__ k,   // (B,H,W,D)
                 const bf16_t* __restrict__ vT,  // (B,H,D,W)
                 const bf16_t* __restrict__ erT, // (W,D)
                 bf16_t* __restrict__ obf) {     // (B,W,C)
    __shared__ __align__(16) float  sS[16 * 32];      // raw QK tile
    __shared__ __align__(16) float  sT[32 * 16];      // bias tile, sT[j][i]
    __shared__ __align__(16) bf16_t sP[16 * 32];      // probs (A fragment source)
    __shared__ __align__(16) bf16_t sK[2][32 * DD];   // TDM K tiles (rows J.., D wide)
    __shared__ __align__(16) bf16_t sV[2][DD * 32];   // TDM V tiles (rows d, 32 wide)
    __shared__ float sM[16], sL[16], sA[16];

    const int lane = threadIdx.x;
    const int nI   = WW / 16;                 // 64 row blocks
    const int i0   = (blockIdx.x % nI) * 16;
    const int bh   = blockIdx.x / nI;         // b*H + h
    const int bi   = bh / HH;
    const int hi   = bh % HH;

    const bf16_t* qh = q  + (size_t)bh * WW * DD;
    const bf16_t* kh = k  + (size_t)bh * WW * DD;
    const bf16_t* vh = vT + (size_t)bh * DD * WW;

    const int n  = lane & 15;
    const int lh = lane >> 4;
    const float rscale = 0.03125f;            // 1/sqrt(C)

    // loop-invariant fragments
    v16bf aq0 = load_frag(qh + (size_t)i0 * DD + 0,  DD, lane);
    v16bf aq1 = load_frag(qh + (size_t)i0 * DD + 32, DD, lane);
    v16bf be0 = load_frag(erT + (size_t)i0 * DD + 0,  DD, lane);
    v16bf be1 = load_frag(erT + (size_t)i0 * DD + 32, DD, lane);

    v8f o0 = {}, o1 = {}, o2 = {}, o3 = {};
    if (lane < 16) { sM[lane] = -3.0e38f; sL[lane] = 0.0f; }
    __syncthreads();

    // prologue: TDM-load tiles for J = 0 into buffer 0
    tdm_load_2d((unsigned)(uintptr_t)&sK[0][0], kh,
                /*tile*/ DD, 32, /*tensor*/ DD, WW, /*stride*/ DD);
    tdm_load_2d((unsigned)(uintptr_t)&sV[0][0], vh,
                /*tile*/ 32, DD, /*tensor*/ WW, DD, /*stride*/ WW);

    for (int J = 0; J < WW; J += 32) {
        const int buf = (J >> 5) & 1;
        // issue next tiles while current WMMAs run, then wait for current pair
        if (J + 32 < WW) {
            tdm_load_2d((unsigned)(uintptr_t)&sK[buf ^ 1][0],
                        kh + (size_t)(J + 32) * DD, DD, 32, DD, WW, DD);
            tdm_load_2d((unsigned)(uintptr_t)&sV[buf ^ 1][0],
                        vh + (J + 32), 32, DD, WW, DD, WW);
            __builtin_amdgcn_s_wait_tensorcnt((short)2);
        } else {
            __builtin_amdgcn_s_wait_tensorcnt((short)0);
        }
        __syncthreads();
        const bf16_t* sKb = &sK[buf][0];
        const bf16_t* sVb = &sV[buf][0];

#pragma unroll
        for (int js = 0; js < 32; js += 16) {
            // QK^T sub-tile (16x16), K-dim = D = 64; K fragments from LDS
            v16bf bk0 = load_frag(sKb + (size_t)js * DD + 0,  DD, lane);
            v16bf bk1 = load_frag(sKb + (size_t)js * DD + 32, DD, lane);
            v8f s = {};
            s = wmma_bf16(aq0, bk0, s);
            s = wmma_bf16(aq1, bk1, s);
            // rel-pos bias sub-tile: T = Q_{J+js} @ erT_{i0}  (then used transposed)
            v16bf aj0 = load_frag(qh + (size_t)(J + js) * DD + 0,  DD, lane);
            v16bf aj1 = load_frag(qh + (size_t)(J + js) * DD + 32, DD, lane);
            v8f t = {};
            t = wmma_bf16(aj0, be0, t);
            t = wmma_bf16(aj1, be1, t);
#pragma unroll
            for (int r = 0; r < 8; ++r) {
                int m = r + 8 * lh;
                sS[m * 32 + js + n]   = s[r];        // sS[i][j]
                sT[(js + m) * 16 + n] = t[r];        // sT[j][i]
            }
        }
        __syncthreads();

        // online softmax over this 16x32 tile: lane -> (row = n, col half = lh)
        int row = n, c0 = lh * 16;
        float vals[16];
        float mx = -3.0e38f;
#pragma unroll
        for (int c = 0; c < 16; ++c) {
            int j = c0 + c;
            float vv = (sS[row * 32 + j] + sT[j * 16 + row]) * rscale;
            vals[c] = vv;
            mx = fmaxf(mx, vv);
        }
        mx = fmaxf(mx, __shfl_xor(mx, 16));
        float mold = sM[row];
        float mnew = fmaxf(mold, mx);
        float lsum = 0.0f;
#pragma unroll
        for (int c = 0; c < 16; ++c) {
            float p = __expf(vals[c] - mnew);
            sP[row * 32 + c0 + c] = f2bf(p);
            lsum += p;
        }
        lsum += __shfl_xor(lsum, 16);
        if (lh == 0) {
            float alpha = __expf(mold - mnew);
            sA[row] = alpha;
            sM[row] = mnew;
            sL[row] = sL[row] * alpha + lsum;
        }
        __syncthreads();

        // rescale O and accumulate P @ V  (K = 32 columns of probs, V from LDS)
        float al[8];
#pragma unroll
        for (int r = 0; r < 8; ++r) al[r] = sA[r + 8 * lh];
        v16bf pa  = load_frag(sP, 32, lane);
        v16bf bv0 = load_frag(sVb + (size_t)(0 * 16) * 32, 32, lane);
        v16bf bv1 = load_frag(sVb + (size_t)(1 * 16) * 32, 32, lane);
        v16bf bv2 = load_frag(sVb + (size_t)(2 * 16) * 32, 32, lane);
        v16bf bv3 = load_frag(sVb + (size_t)(3 * 16) * 32, 32, lane);
#pragma unroll
        for (int r = 0; r < 8; ++r) {
            o0[r] *= al[r]; o1[r] *= al[r]; o2[r] *= al[r]; o3[r] *= al[r];
        }
        o0 = wmma_bf16(pa, bv0, o0);
        o1 = wmma_bf16(pa, bv1, o1);
        o2 = wmma_bf16(pa, bv2, o2);
        o3 = wmma_bf16(pa, bv3, o3);
        __syncthreads();
    }

    // normalize and emit into (B, W, C) bf16 for the output projection
    float linv[8];
#pragma unroll
    for (int r = 0; r < 8; ++r) linv[r] = 1.0f / sL[r + 8 * lh];
#pragma unroll
    for (int r = 0; r < 8; ++r) {
        int m = r + 8 * lh;
        size_t base = ((size_t)bi * WW + (i0 + m)) * CC + hi * DD;
        obf[base + 0 * 16 + n]  = f2bf(o0[r] * linv[r]);
        obf[base + 1 * 16 + n]  = f2bf(o1[r] * linv[r]);
        obf[base + 2 * 16 + n]  = f2bf(o2[r] * linv[r]);
        obf[base + 3 * 16 + n]  = f2bf(o3[r] * linv[r]);
    }
}

// ---------------------------------------------------------------------------
// host side
// ---------------------------------------------------------------------------
extern "C" void kernel_launch(void* const* d_in, const int* in_sizes, int n_in,
                              void* d_out, int out_size, void* d_ws, size_t ws_size,
                              hipStream_t stream) {
    const float* x   = (const float*)d_in[0];
    const float* qW  = (const float*)d_in[1];
    const float* qb  = (const float*)d_in[2];
    const float* qcw = (const float*)d_in[3];
    const float* qcb = (const float*)d_in[4];
    const float* kW  = (const float*)d_in[5];
    const float* kb  = (const float*)d_in[6];
    const float* kcw = (const float*)d_in[7];
    const float* kcb = (const float*)d_in[8];
    const float* vW  = (const float*)d_in[9];
    const float* vb  = (const float*)d_in[10];
    const float* vcw = (const float*)d_in[11];
    const float* vcb = (const float*)d_in[12];
    const float* oW  = (const float*)d_in[13];
    const float* ob  = (const float*)d_in[14];
    const float* er  = (const float*)d_in[15];
    float* out = (float*)d_out;

    // workspace carve-up (256B aligned)
    char* ws = (char*)d_ws;
    size_t off = 0;
    auto alloc = [&](size_t bytes) -> void* {
        void* p = ws + off;
        off += (bytes + 255) & ~(size_t)255;
        return p;
    };
    bf16_t* xbf  = (bf16_t*)alloc((size_t)NN * CC * 2);
    bf16_t* qWb  = (bf16_t*)alloc((size_t)CC * CC * 2);
    bf16_t* kWb  = (bf16_t*)alloc((size_t)CC * CC * 2);
    bf16_t* vWb  = (bf16_t*)alloc((size_t)CC * CC * 2);
    bf16_t* oWb  = (bf16_t*)alloc((size_t)CC * CC * 2);
    bf16_t* erTb = (bf16_t*)alloc((size_t)WW * DD * 2);
    float*  lin  = (float*) alloc((size_t)NN * CC * 4);
    bf16_t* qbf  = (bf16_t*)alloc((size_t)NN * CC * 2);
    bf16_t* kbf  = (bf16_t*)alloc((size_t)NN * CC * 2);
    bf16_t* vTb  = (bf16_t*)alloc((size_t)NN * CC * 2);
    bf16_t* obf  = (bf16_t*)alloc((size_t)NN * CC * 2);

    const int T = 256;
    // casts
    cast_f32_bf16_kernel<<<(NN * CC + T - 1) / T, T, 0, stream>>>(x,  xbf, NN * CC);
    cast_f32_bf16_kernel<<<(CC * CC + T - 1) / T, T, 0, stream>>>(qW, qWb, CC * CC);
    cast_f32_bf16_kernel<<<(CC * CC + T - 1) / T, T, 0, stream>>>(kW, kWb, CC * CC);
    cast_f32_bf16_kernel<<<(CC * CC + T - 1) / T, T, 0, stream>>>(vW, vWb, CC * CC);
    cast_f32_bf16_kernel<<<(CC * CC + T - 1) / T, T, 0, stream>>>(oW, oWb, CC * CC);
    transpose_er_kernel<<<(DD * WW + T - 1) / T, T, 0, stream>>>(er, erTb);

    const int tiles  = (NN / 64) * (CC / 64);   // 1024 waves
    const int gblks  = tiles / 8;               // 8 waves (256 thr) per block
    const int eblks  = (NN * CC + T - 1) / T;

    // q
    gemm_bf16_kernel<<<gblks, T, 0, stream>>>(xbf, qWb, qb, lin, NN, CC, CC);
    conv_dw_kernel  <<<eblks, T, 0, stream>>>(lin, qcw, qcb, qbf, 0);
    // k
    gemm_bf16_kernel<<<gblks, T, 0, stream>>>(xbf, kWb, kb, lin, NN, CC, CC);
    conv_dw_kernel  <<<eblks, T, 0, stream>>>(lin, kcw, kcb, kbf, 0);
    // v (transposed per-head layout)
    gemm_bf16_kernel<<<gblks, T, 0, stream>>>(xbf, vWb, vb, lin, NN, CC, CC);
    conv_dw_kernel  <<<eblks, T, 0, stream>>>(lin, vcw, vcb, vTb, 1);

    // attention: one wave per (b, h, 16-row block)
    attn_kernel<<<BB * HH * (WW / 16), 32, 0, stream>>>(qbf, kbf, vTb, erTb, obf);

    // output projection -> fp32 result
    gemm_bf16_kernel<<<gblks, T, 0, stream>>>(obf, oWb, ob, out, NN, CC, CC);
}
